// Bidirectional_GRU_Decoder_87995289960799
// MI455X (gfx1250) — matrix operand order repaired and checked by reading, not compiled, not run on hardware
//
#include <hip/hip_runtime.h>
#include <hip/hip_bf16.h>
#include <math.h>

#define BB 64
#define SS 512
#define II 512
#define HH 512
#define G3H 1536
#define NLAYER 2

typedef __attribute__((ext_vector_type(16))) __bf16    v16bf;
typedef __attribute__((ext_vector_type(8)))  float     v8f;
typedef __attribute__((ext_vector_type(8)))  unsigned  v8u;

union FragU { v16bf h; v8u u; };

__device__ inline v8f wmma_bf16(v16bf a, v16bf b, v8f c) {
  return __builtin_amdgcn_wmma_f32_16x16x32_bf16(false, a, false, b, (short)0, c,
                                                 false, false);
}

// 16-bit A matrix fragment, 16x32 (MxK). ISA layout:
// lane m=lane&15, half=lane>>4; vgpr i holds K pair k = 2i + 8*half (+8 if i>=4).
__device__ inline v16bf load_frag_a(const __bf16* base, long long rowBase, int rowStep,
                                    int ld, int kb, int lane) {
  int m = lane & 15, half = lane >> 4;
  long long row = rowBase + (long long)rowStep * m;
  const __bf16* rp = base + row * (long long)ld + kb;
  FragU f;
#pragma unroll
  for (int i = 0; i < 8; ++i) {
    int k = 2 * i + 8 * half + ((i >= 4) ? 8 : 0);
    f.u[i] = *(const unsigned*)(rp + k);
  }
  return f.h;
}

// 16-bit B matrix fragment, 32x16 (KxN), sourced from row-major W[N][K] (B = W^T).
// lane n=lane&15, half=lane>>4; vgpr i holds K pair k = 2i + 8*half (+8 if i>=4).
__device__ inline v16bf load_frag_b(const __bf16* w, int ncolBase, int ld, int kb,
                                    int lane) {
  int n = lane & 15, half = lane >> 4;
  const __bf16* rp = w + (long long)(ncolBase + n) * ld + kb;
  FragU f;
#pragma unroll
  for (int i = 0; i < 8; ++i) f.u[i] = *(const unsigned*)(rp + 2 * i + 8 * half + ((i >= 4) ? 8 : 0));
  return f.h;
}

__global__ void f32_to_bf16_kernel(const float* __restrict__ in, __bf16* __restrict__ out,
                                   long long n) {
  long long i = (long long)blockIdx.x * blockDim.x + threadIdx.x;
  long long stride = (long long)gridDim.x * blockDim.x;
  for (; i < n; i += stride) out[i] = (__bf16)in[i];
}

__global__ void init_ctr_kernel(unsigned* ctr) {
  if (threadIdx.x < 8) ctr[threadIdx.x] = 0u;
}

// gi = A(bf16)[M=B*S, K] x W(bf16)[N=3H, K]^T + bias.
// Output is f32, *t-major*: gi[(t*BB + b) * 3H + col].
// rev!=0: A row for output row (b,t) is (b, S-1-t)  (backward direction, layer 0).
// K must be a multiple of 64 (it is 512 in all uses).
__global__ __launch_bounds__(256) void gemm_gi_kernel(
    const __bf16* __restrict__ A, const __bf16* __restrict__ W,
    const float* __restrict__ bias, float* __restrict__ out, int K, int rev) {
  int lane = threadIdx.x & 31;
  int wave = threadIdx.x >> 5;
  int mBase = blockIdx.y * 128 + (wave >> 2) * 64;   // 2 waves along M
  int nBase = blockIdx.x * 128 + (wave & 3) * 32;    // 4 waves along N

  v8f zv = {};
  v8f acc[4][2];
#pragma unroll
  for (int mt = 0; mt < 4; ++mt) { acc[mt][0] = zv; acc[mt][1] = zv; }

  // per-mt A row base (handles time reversal), constant across k
  long long rb[4];
  int stp = rev ? -1 : 1;
#pragma unroll
  for (int mt = 0; mt < 4; ++mt) {
    int mRow = mBase + mt * 16;
    if (rev) { int b = mRow >> 9; rb[mt] = (long long)(2 * b + 1) * SS - 1 - mRow; }
    else     { rb[mt] = mRow; }
  }

  // ping-pong software pipeline: two fragment sets swap roles, no register copies
  v16bf b0a = load_frag_b(W, nBase, K, 0, lane);
  v16bf b1a = load_frag_b(W, nBase + 16, K, 0, lane);
  v16bf aa[4];
#pragma unroll
  for (int mt = 0; mt < 4; ++mt) aa[mt] = load_frag_a(A, rb[mt], stp, K, 0, lane);

  for (int kb = 0; kb < K; kb += 64) {
    // load set B at kb+32 (always in range: K % 64 == 0)
    v16bf b0b = load_frag_b(W, nBase, K, kb + 32, lane);
    v16bf b1b = load_frag_b(W, nBase + 16, K, kb + 32, lane);
    v16bf ab[4];
#pragma unroll
    for (int mt = 0; mt < 4; ++mt) ab[mt] = load_frag_a(A, rb[mt], stp, K, kb + 32, lane);

#pragma unroll
    for (int mt = 0; mt < 4; ++mt) {
      acc[mt][0] = wmma_bf16(aa[mt], b0a, acc[mt][0]);
      acc[mt][1] = wmma_bf16(aa[mt], b1a, acc[mt][1]);
    }

    if (kb + 64 < K) {  // preload set A at kb+64 (skipped on final iteration)
      b0a = load_frag_b(W, nBase, K, kb + 64, lane);
      b1a = load_frag_b(W, nBase + 16, K, kb + 64, lane);
#pragma unroll
      for (int mt = 0; mt < 4; ++mt) aa[mt] = load_frag_a(A, rb[mt], stp, K, kb + 64, lane);
    }

#pragma unroll
    for (int mt = 0; mt < 4; ++mt) {
      acc[mt][0] = wmma_bf16(ab[mt], b0b, acc[mt][0]);
      acc[mt][1] = wmma_bf16(ab[mt], b1b, acc[mt][1]);
    }
  }

  int half = lane >> 4, nn = lane & 15;
#pragma unroll
  for (int mt = 0; mt < 4; ++mt)
#pragma unroll
    for (int nt = 0; nt < 2; ++nt) {
      int col = nBase + nt * 16 + nn;
      float bs = bias[col];
#pragma unroll
      for (int r = 0; r < 8; ++r) {
        int row = mBase + mt * 16 + r + 8 * half;
        int b = row >> 9;        // / S
        int t = row & (SS - 1);  // % S
        out[((size_t)t * BB + b) * G3H + col] = acc[mt][nt][r] + bs;
      }
    }
}

__device__ inline void grid_sync_dir(unsigned* ctr, unsigned target) {
  __syncthreads();
  __threadfence();
  if (threadIdx.x == 0) {
    atomicAdd(ctr, 1u);
    while (*(volatile unsigned*)ctr < target) { __builtin_amdgcn_s_sleep(1); }
  }
  __syncthreads();
  __threadfence();
}

// Persistent GRU scan. Grid (8, 2): 8 blocks x 8 waves = 64 waves per direction.
// Wave w of block x handles column slice (x*4 + (w>>1)) and batch half (w&1):
// a 32x16 h-patch held in registers for all S steps (2 M-tiles x 3 gates = 6
// WMMA accumulators). gi is t-major (one contiguous 384KB block per step).
__global__ __launch_bounds__(256) void gru_scan_kernel(
    const float* __restrict__ gi_f, const float* __restrict__ gi_b,
    const __bf16* __restrict__ whh_f, const __bf16* __restrict__ whh_b,
    const float* __restrict__ bhh_f, const float* __restrict__ bhh_b,
    const float* __restrict__ encoder_h,
    __bf16* __restrict__ hbuf_f, __bf16* __restrict__ hbuf_b,
    __bf16* __restrict__ stream_f, __bf16* __restrict__ stream_b,
    float* __restrict__ outF32, float* __restrict__ hOut,
    unsigned* __restrict__ ctrs, int layer) {
  const int dir = blockIdx.y;
  const float* gi    = dir ? gi_b : gi_f;
  const __bf16* Whh  = dir ? whh_b : whh_f;
  const float* bhh   = dir ? bhh_b : bhh_f;
  __bf16* hbuf       = dir ? hbuf_b : hbuf_f;
  __bf16* stream     = dir ? stream_b : stream_f;
  unsigned* ctr      = ctrs + layer * 2 + dir;

  int lane  = threadIdx.x & 31;
  int wave  = threadIdx.x >> 5;        // 0..7
  int c     = (blockIdx.x * 4 + (wave >> 1)) * 16;  // column slice base in [0,H)
  int mhalf = wave & 1;                // batch half: rows [mhalf*32, mhalf*32+32)
  int nn    = lane & 15;
  int half  = lane >> 4;
  int col   = c + nn;
  int mt0   = mhalf * 2;               // first of this wave's 2 M-tiles

  // h patch registers, C-tile layout: hreg[mt][r] = h[b = (mt0+mt)*16 + r + 8*half][col]
  v8f hreg[2];
#pragma unroll
  for (int mt = 0; mt < 2; ++mt)
#pragma unroll
    for (int r = 0; r < 8; ++r) {
      int b = (mt0 + mt) * 16 + r + 8 * half;
      hreg[mt][r] = encoder_h[(size_t)b * (2 * HH) + dir * HH + col];
    }
  // seed parity-0 bf16 h buffer
#pragma unroll
  for (int mt = 0; mt < 2; ++mt)
#pragma unroll
    for (int r = 0; r < 8; ++r) {
      int b = (mt0 + mt) * 16 + r + 8 * half;
      hbuf[(size_t)b * HH + col] = (__bf16)hreg[mt][r];
    }

  const float bhr = bhh[col], bhz = bhh[HH + col], bhn = bhh[2 * HH + col];
  unsigned bar = 1;
  grid_sync_dir(ctr, bar * 8); ++bar;

  v8f zv = {};
  for (int t = 0; t < SS; ++t) {
    const __bf16* hread  = hbuf + (size_t)(t & 1) * (BB * HH);
    __bf16*       hwrite = hbuf + (size_t)((t + 1) & 1) * (BB * HH);

    v8f acc[3][2];
#pragma unroll
    for (int g = 0; g < 3; ++g)
#pragma unroll
      for (int mt = 0; mt < 2; ++mt) acc[g][mt] = zv;

    // ping-pong pipelined k loop (HH % 64 == 0)
    v16bf b0a = load_frag_b(Whh, 0 * HH + c, HH, 0, lane);
    v16bf b1a = load_frag_b(Whh, 1 * HH + c, HH, 0, lane);
    v16bf b2a = load_frag_b(Whh, 2 * HH + c, HH, 0, lane);
    v16bf a0a = load_frag_a(hread, (mt0 + 0) * 16, 1, HH, 0, lane);
    v16bf a1a = load_frag_a(hread, (mt0 + 1) * 16, 1, HH, 0, lane);
    for (int kb = 0; kb < HH; kb += 64) {
      v16bf b0b = load_frag_b(Whh, 0 * HH + c, HH, kb + 32, lane);
      v16bf b1b = load_frag_b(Whh, 1 * HH + c, HH, kb + 32, lane);
      v16bf b2b = load_frag_b(Whh, 2 * HH + c, HH, kb + 32, lane);
      v16bf a0b = load_frag_a(hread, (mt0 + 0) * 16, 1, HH, kb + 32, lane);
      v16bf a1b = load_frag_a(hread, (mt0 + 1) * 16, 1, HH, kb + 32, lane);

      acc[0][0] = wmma_bf16(a0a, b0a, acc[0][0]);
      acc[1][0] = wmma_bf16(a0a, b1a, acc[1][0]);
      acc[2][0] = wmma_bf16(a0a, b2a, acc[2][0]);
      acc[0][1] = wmma_bf16(a1a, b0a, acc[0][1]);
      acc[1][1] = wmma_bf16(a1a, b1a, acc[1][1]);
      acc[2][1] = wmma_bf16(a1a, b2a, acc[2][1]);

      if (kb + 64 < HH) {
        b0a = load_frag_b(Whh, 0 * HH + c, HH, kb + 64, lane);
        b1a = load_frag_b(Whh, 1 * HH + c, HH, kb + 64, lane);
        b2a = load_frag_b(Whh, 2 * HH + c, HH, kb + 64, lane);
        a0a = load_frag_a(hread, (mt0 + 0) * 16, 1, HH, kb + 64, lane);
        a1a = load_frag_a(hread, (mt0 + 1) * 16, 1, HH, kb + 64, lane);
      }

      acc[0][0] = wmma_bf16(a0b, b0b, acc[0][0]);
      acc[1][0] = wmma_bf16(a0b, b1b, acc[1][0]);
      acc[2][0] = wmma_bf16(a0b, b2b, acc[2][0]);
      acc[0][1] = wmma_bf16(a1b, b0b, acc[0][1]);
      acc[1][1] = wmma_bf16(a1b, b1b, acc[1][1]);
      acc[2][1] = wmma_bf16(a1b, b2b, acc[2][1]);
    }

    // warm caches for next timestep's contiguous gi block (global_prefetch_b8)
    if (t + 1 < SS) {
      const float* nxt = gi + (size_t)(t + 1) * (BB * G3H);
      __builtin_prefetch(nxt + (size_t)threadIdx.x * 384, 0, 0);
    }

#pragma unroll
    for (int mt = 0; mt < 2; ++mt) {
#pragma unroll
      for (int r = 0; r < 8; ++r) {
        int b = (mt0 + mt) * 16 + r + 8 * half;
        const float* girow = gi + ((size_t)t * BB + b) * G3H;
        float ir  = girow[col];
        float iz  = girow[HH + col];
        float inp = girow[2 * HH + col];
        float hr = acc[0][mt][r] + bhr;
        float hz = acc[1][mt][r] + bhz;
        float hn = acc[2][mt][r] + bhn;
        float rg = 1.f / (1.f + __expf(-(ir + hr)));
        float zg = 1.f / (1.f + __expf(-(iz + hz)));
        float ng = tanhf(inp + rg * hn);
        float hp = hreg[mt][r];
        float hv = (1.f - zg) * ng + zg * hp;
        hreg[mt][r] = hv;
        hwrite[(size_t)b * HH + col] = (__bf16)hv;
        if (stream) stream[((size_t)b * SS + t) * HH + col] = (__bf16)hv;
        if (outF32) {
          int s = dir ? (SS - 1 - t) : t;
          outF32[((size_t)b * SS + s) * (2 * HH) + dir * HH + col] = hv;
        }
      }
    }
    grid_sync_dir(ctr, bar * 8); ++bar;
  }

  if (hOut) {
#pragma unroll
    for (int mt = 0; mt < 2; ++mt)
#pragma unroll
      for (int r = 0; r < 8; ++r) {
        int b = (mt0 + mt) * 16 + r + 8 * half;
        hOut[(size_t)b * (2 * HH) + dir * HH + col] = hreg[mt][r];
      }
  }
}

extern "C" void kernel_launch(void* const* d_in, const int* in_sizes, int n_in,
                              void* d_out, int out_size, void* d_ws, size_t ws_size,
                              hipStream_t stream) {
  (void)in_sizes; (void)n_in; (void)out_size; (void)ws_size;
  const float* x     = (const float*)d_in[0];
  const float* enc_h = (const float*)d_in[1];
  const float* Wih_f = (const float*)d_in[2];
  const float* Whh_f = (const float*)d_in[3];
  const float* bih_f = (const float*)d_in[4];
  const float* bhh_f = (const float*)d_in[5];
  const float* Wih_b = (const float*)d_in[6];
  const float* Whh_b = (const float*)d_in[7];
  const float* bih_b = (const float*)d_in[8];
  const float* bhh_b = (const float*)d_in[9];
  float* out  = (float*)d_out;
  float* hOut = out + (size_t)BB * SS * 2 * HH;

  char* ws = (char*)d_ws;
  size_t off = 0;
  auto alloc = [&](size_t bytes) -> void* {
    void* p = ws + off;
    off += (bytes + 255) & ~(size_t)255;
    return p;
  };
  __bf16* xbf  = (__bf16*)alloc((size_t)BB * SS * II * 2);
  __bf16* wihf = (__bf16*)alloc((size_t)NLAYER * G3H * II * 2);
  __bf16* whhf = (__bf16*)alloc((size_t)NLAYER * G3H * HH * 2);
  __bf16* wihb = (__bf16*)alloc((size_t)NLAYER * G3H * II * 2);
  __bf16* whhb = (__bf16*)alloc((size_t)NLAYER * G3H * HH * 2);
  float*  gif  = (float*)alloc((size_t)BB * SS * G3H * 4);
  float*  gib  = (float*)alloc((size_t)BB * SS * G3H * 4);
  __bf16* stf  = (__bf16*)alloc((size_t)BB * SS * HH * 2);
  __bf16* stb  = (__bf16*)alloc((size_t)BB * SS * HH * 2);
  __bf16* hbf  = (__bf16*)alloc((size_t)2 * BB * HH * 2);
  __bf16* hbb  = (__bf16*)alloc((size_t)2 * BB * HH * 2);
  unsigned* ctrs = (unsigned*)alloc(256);

  // reset barrier counters every launch (deterministic across graph replays)
  init_ctr_kernel<<<1, 32, 0, stream>>>(ctrs);

  // f32 -> bf16 converts
  long long nx = (long long)BB * SS * II;
  long long nw = (long long)NLAYER * G3H * II;
  f32_to_bf16_kernel<<<2048, 256, 0, stream>>>(x, xbf, nx);
  f32_to_bf16_kernel<<<512, 256, 0, stream>>>(Wih_f, wihf, nw);
  f32_to_bf16_kernel<<<512, 256, 0, stream>>>(Whh_f, whhf, nw);
  f32_to_bf16_kernel<<<512, 256, 0, stream>>>(Wih_b, wihb, nw);
  f32_to_bf16_kernel<<<512, 256, 0, stream>>>(Whh_b, whhb, nw);

  dim3 ggrid(G3H / 128, (BB * SS) / 128);  // (12, 256)
  dim3 sgrid(8, 2);

  // layer 0: input projections (backward reads x time-reversed), then scan
  gemm_gi_kernel<<<ggrid, 256, 0, stream>>>(xbf, wihf, bih_f, gif, II, 0);
  gemm_gi_kernel<<<ggrid, 256, 0, stream>>>(xbf, wihb, bih_b, gib, II, 1);
  gru_scan_kernel<<<sgrid, 256, 0, stream>>>(gif, gib, whhf, whhb, bhh_f, bhh_b, enc_h,
                                             hbf, hbb, stf, stb, nullptr, nullptr,
                                             ctrs, 0);

  // layer 1: projections from layer-0 streams (already in scan-time order), then scan
  gemm_gi_kernel<<<ggrid, 256, 0, stream>>>(stf, wihf + (size_t)G3H * II,
                                            bih_f + G3H, gif, HH, 0);
  gemm_gi_kernel<<<ggrid, 256, 0, stream>>>(stb, wihb + (size_t)G3H * II,
                                            bih_b + G3H, gib, HH, 0);
  gru_scan_kernel<<<sgrid, 256, 0, stream>>>(gif, gib,
                                             whhf + (size_t)G3H * HH,
                                             whhb + (size_t)G3H * HH,
                                             bhh_f + G3H, bhh_b + G3H, enc_h,
                                             hbf, hbb, nullptr, nullptr,
                                             out, hOut, ctrs, 1);
}
